// CrossAttention_25400436588552
// MI455X (gfx1250) — matrix operand order repaired and checked by reading, not compiled, not run on hardware
//
#include <hip/hip_runtime.h>

// ---------------- CDNA5 WMMA types ----------------
typedef __bf16 bf16;
typedef __attribute__((ext_vector_type(16))) __bf16 v16bf;
typedef __attribute__((ext_vector_type(8)))  __bf16 v8bf;
typedef __attribute__((ext_vector_type(8)))  float  v8f;

#define WMMA_BF16(A, B, C) \
    __builtin_amdgcn_wmma_f32_16x16x32_bf16(false, (A), false, (B), (short)0, (C), false, false)

static __device__ __forceinline__ v16bf cat8(v8bf lo, v8bf hi) {
    v16bf r;
#pragma unroll
    for (int e = 0; e < 8; ++e) { r[e] = lo[e]; r[e + 8] = hi[e]; }
    return r;
}

// Problem constants (from reference)
#define BATCH   2
#define NQ      2048
#define NK      2048
#define QDIM    1024
#define CDIM    1024
#define HEADS   8
#define DHEAD   64
#define INNER   512          // HEADS*DHEAD
#define SCALE   0.125f       // 64^-0.5

// ---------------------------------------------------------------------------
// Kernel 0: transpose + fp32 -> bf16 convert.  src is R x C row-major,
// dst is C x R row-major.  C is a power of two -> shift/mask, no divide.
// ---------------------------------------------------------------------------
__global__ void transpose_cvt(const float* __restrict__ src,
                              bf16* __restrict__ dst, int R, int lgC) {
    const int idx = blockIdx.x * blockDim.x + threadIdx.x;
    const int r = idx >> lgC;
    const int c = idx & ((1 << lgC) - 1);
    if (r >= R) return;
    dst[(size_t)c * R + r] = (__bf16)src[idx];
}

// ---------------------------------------------------------------------------
// Kernel 1: QKV projections.  blockIdx.z = 0(Q) / 1(K) / 2(V).
// Each wave computes a 32x64 output tile (2 A-frags x 4 B-frags = 8 WMMAs
// per 32-wide K step); 8 waves / block -> 256 rows per block.
// Q -> [B][H][N][64], K -> [B][H][M][64], V -> transposed [B][H][64][M].
// ---------------------------------------------------------------------------
__global__ __launch_bounds__(256) void qkv_gemm(
    const float* __restrict__ x,    // [B*N][1024]
    const float* __restrict__ ctx,  // [B*M][1024]
    const bf16* __restrict__ WqT,   // [512][1024]
    const bf16* __restrict__ WkT,   // [512][1024]
    const bf16* __restrict__ WvT,   // [512][1024]
    bf16* __restrict__ qb, bf16* __restrict__ kb, bf16* __restrict__ vtb) {
    const int which = blockIdx.z;
    const float* A  = (which == 0) ? x : ctx;
    const bf16*  WT = (which == 0) ? WqT : (which == 1) ? WkT : WvT;

    const int lane = threadIdx.x & 31;
    const int w    = threadIdx.x >> 5;
    const int r    = lane & 15;
    const int hi   = (lane >= 16) ? 1 : 0;
    const int da   = hi * 8;
    const int rowBase = blockIdx.x * 256 + w * 32;   // global row (b*2048 + i)
    const int colBase = blockIdx.y * 64;             // inner index n

    v8f acc0[4] = {}, acc1[4] = {};
    const float* arow0 = A + (size_t)(rowBase + r) * QDIM;
    const float* arow1 = arow0 + (size_t)16 * QDIM;

    for (int k0 = 0; k0 < QDIM; k0 += 32) {
        v16bf a0, a1;
        const int ka = k0 + hi * 16;
#pragma unroll
        for (int e = 0; e < 8; ++e) {
            a0[e]     = (__bf16)arow0[ka + e];
            a0[e + 8] = (__bf16)arow0[ka + 8 + e];
            a1[e]     = (__bf16)arow1[ka + e];
            a1[e + 8] = (__bf16)arow1[ka + 8 + e];
        }
#pragma unroll
        for (int t = 0; t < 4; ++t) {
            const bf16* bp = WT + (size_t)(colBase + t * 16 + r) * QDIM + k0 + hi * 16;
            v16bf bfrag = *(const v16bf*)bp;   // contiguous 32B per lane
            acc0[t] = WMMA_BF16(a0, bfrag, acc0[t]);
            acc1[t] = WMMA_BF16(a1, bfrag, acc1[t]);
        }
    }

    // Branch-free epilogue: pick destination pointer + element stride once.
    const int b  = rowBase >> 11;          // /2048
    const int i  = rowBase & 2047;
    const int bh = b * HEADS;              // + h added per tile
#pragma unroll
    for (int t = 0; t < 4; ++t) {
        const int n = colBase + t * 16 + r;
        const int h = n >> 6, d = n & 63;
        bf16*  dst;
        size_t es;                         // stride between consecutive rows
        if (which == 2) {                  // V: transposed [bh][d][row]
            dst = vtb + ((size_t)((bh + h) * DHEAD + d)) * NK + i + da;
            es  = 1;
        } else {                           // Q/K: [bh][row][d]
            bf16* base01 = (which == 0) ? qb : kb;
            dst = base01 + ((size_t)(bh + h) * NQ + i + da) * DHEAD + d;
            es  = DHEAD;
        }
#pragma unroll
        for (int e = 0; e < 8; ++e) {
            dst[(size_t)e * es]                   = (__bf16)acc0[t][e];
            dst[(size_t)(e + 16) * es]            = (__bf16)acc1[t][e];
        }
    }
}

// ---------------------------------------------------------------------------
// Kernel 2: flash-style attention.  One wave = 16 queries of one (b,h);
// streams over all 2048 keys in 32-key panels with online softmax.
// ---------------------------------------------------------------------------
#define PSTRIDE 40   // padded LDS row stride (bf16) -> rows stay 16B aligned
__global__ __launch_bounds__(256) void attn_kernel(
    const bf16* __restrict__ qb,   // [B][H][N][64]
    const bf16* __restrict__ kb,   // [B][H][M][64]
    const bf16* __restrict__ vtb,  // [B][H][64][M]
    bf16* __restrict__ attO) {     // [B][N][512]  (b,i,h,d)
    __shared__ __align__(32) __bf16 lsP[8][16][PSTRIDE];

    const int lane = threadIdx.x & 31;
    const int w    = threadIdx.x >> 5;
    const int r    = lane & 15;
    const int hi   = (lane >= 16) ? 1 : 0;
    const int da   = hi * 8;
    const int b    = blockIdx.z;
    const int h    = blockIdx.y;
    const int q0   = (blockIdx.x * 8 + w) * 16;

    const bf16* qbase = qb  + ((size_t)(b * HEADS + h) * NQ + q0) * DHEAD;
    const bf16* kbase = kb  + (size_t)(b * HEADS + h) * NK * DHEAD;
    const bf16* vbase = vtb + (size_t)(b * HEADS + h) * DHEAD * NK;

    // Q A-fragments (held in registers), vector 16B loads
    const bf16* qrow = qbase + r * DHEAD;
    const v16bf qa0 = cat8(*(const v8bf*)(qrow + hi * 16),      *(const v8bf*)(qrow + hi * 16 + 8));
    const v16bf qa1 = cat8(*(const v8bf*)(qrow + 32 + hi * 16), *(const v8bf*)(qrow + 40 + hi * 16));

    float mrow[8], lrow[8];
    v8f o[4] = {};
#pragma unroll
    for (int e = 0; e < 8; ++e) { mrow[e] = -3.0e38f; lrow[e] = 0.f; }

    for (int j0 = 0; j0 < NK; j0 += 32) {
        // prefetch next K panel (global_prefetch_b8 on gfx1250)
        if (j0 + 32 < NK) {
            __builtin_prefetch(kbase + (size_t)(j0 + 32 + r) * DHEAD + hi * 16, 0, 0);
            __builtin_prefetch(vbase + (size_t)r * NK + j0 + 32 + hi * 16, 0, 0);
        }
        // ---- scores: two 16x16 tiles, K-contraction = 64 (2 WMMAs each) ----
        v8f s0 = {}, s1 = {};
        {
            const bf16* kc = kbase + (size_t)(j0 + r) * DHEAD + hi * 16;
            s0 = WMMA_BF16(qa0, *(const v16bf*)kc, s0);
            s0 = WMMA_BF16(qa1, *(const v16bf*)(kc + 32), s0);
        }
        {
            const bf16* kc = kbase + (size_t)(j0 + 16 + r) * DHEAD + hi * 16;
            s1 = WMMA_BF16(qa0, *(const v16bf*)kc, s1);
            s1 = WMMA_BF16(qa1, *(const v16bf*)(kc + 32), s1);
        }
        // ---- online softmax over the 32-key panel ----
        v8f p0, p1;
#pragma unroll
        for (int e = 0; e < 8; ++e) {
            float a0 = s0[e] * SCALE;
            float a1 = s1[e] * SCALE;
            float mx = fmaxf(a0, a1);
            mx = fmaxf(mx, __shfl_xor(mx, 1));
            mx = fmaxf(mx, __shfl_xor(mx, 2));
            mx = fmaxf(mx, __shfl_xor(mx, 4));
            mx = fmaxf(mx, __shfl_xor(mx, 8));
            const float mnew  = fmaxf(mrow[e], mx);
            const float alpha = __expf(mrow[e] - mnew);
            const float e0 = __expf(a0 - mnew);
            const float e1 = __expf(a1 - mnew);
            float sum = e0 + e1;
            sum += __shfl_xor(sum, 1);
            sum += __shfl_xor(sum, 2);
            sum += __shfl_xor(sum, 4);
            sum += __shfl_xor(sum, 8);
            lrow[e] = lrow[e] * alpha + sum;
            mrow[e] = mnew;
            p0[e] = e0;
            p1[e] = e1;
#pragma unroll
            for (int t = 0; t < 4; ++t) o[t][e] = o[t][e] * alpha;
        }
        // ---- C-layout -> A-layout flip through wave-private LDS ----
#pragma unroll
        for (int e = 0; e < 8; ++e) {
            lsP[w][e + da][r]      = (__bf16)p0[e];
            lsP[w][e + da][16 + r] = (__bf16)p1[e];
        }
        asm volatile("s_wait_dscnt 0" ::: "memory");
        const __bf16* prow = &lsP[w][r][0];
        const v16bf pa = cat8(*(const v8bf*)(prow + hi * 16), *(const v8bf*)(prow + hi * 16 + 8));
        // ---- O += P @ V (4 dim-tiles; V^T rows are contiguous) ----
#pragma unroll
        for (int t = 0; t < 4; ++t) {
            const bf16* vc = vbase + (size_t)(t * 16 + r) * NK + j0 + hi * 16;
            o[t] = WMMA_BF16(pa, *(const v16bf*)vc, o[t]);
        }
        asm volatile("s_wait_dscnt 0" ::: "memory");  // WAR guard on lsP reuse
    }

    // ---- epilogue: O /= l, store bf16 into [b][i][h*64+d] ----
#pragma unroll
    for (int e = 0; e < 8; ++e) {
        const float inv = 1.0f / lrow[e];
        const int row = q0 + e + da;
        bf16* dst = attO + (size_t)(b * NQ + row) * INNER + h * 64 + r;
#pragma unroll
        for (int t = 0; t < 4; ++t) dst[t * 16] = (__bf16)(o[t][e] * inv);
    }
}

// ---------------------------------------------------------------------------
// Kernel 3: out = attO(bf16) @ Wo + bo, fp32 output.  32x64 tile per wave.
// ---------------------------------------------------------------------------
__global__ __launch_bounds__(256) void out_gemm(
    const bf16* __restrict__ attO,  // [4096][512]
    const bf16* __restrict__ WoT,   // [1024][512]
    const float* __restrict__ bo,   // [1024]
    float* __restrict__ out) {      // [4096][1024]
    const int lane = threadIdx.x & 31;
    const int w    = threadIdx.x >> 5;
    const int r    = lane & 15;
    const int hi   = (lane >= 16) ? 1 : 0;
    const int da   = hi * 8;
    const int rowBase = blockIdx.x * 256 + w * 32;
    const int colBase = blockIdx.y * 64;

    v8f acc0[4] = {}, acc1[4] = {};
    const bf16* arow0 = attO + (size_t)(rowBase + r) * INNER;
    const bf16* arow1 = arow0 + (size_t)16 * INNER;

    for (int k0 = 0; k0 < INNER; k0 += 32) {
        const int ka = k0 + hi * 16;
        const v16bf a0 = cat8(*(const v8bf*)(arow0 + ka), *(const v8bf*)(arow0 + ka + 8));
        const v16bf a1 = cat8(*(const v8bf*)(arow1 + ka), *(const v8bf*)(arow1 + ka + 8));
#pragma unroll
        for (int t = 0; t < 4; ++t) {
            const bf16* bp = WoT + (size_t)(colBase + t * 16 + r) * INNER + k0 + hi * 16;
            v16bf bfrag = *(const v16bf*)bp;
            acc0[t] = WMMA_BF16(a0, bfrag, acc0[t]);
            acc1[t] = WMMA_BF16(a1, bfrag, acc1[t]);
        }
    }
#pragma unroll
    for (int t = 0; t < 4; ++t) {
        const int n = colBase + t * 16 + r;
        const float bias = bo[n];
        float* dst = out + (size_t)(rowBase + da) * QDIM + n;
#pragma unroll
        for (int e = 0; e < 8; ++e) {
            dst[(size_t)e * QDIM]        = acc0[t][e] + bias;
            dst[(size_t)(e + 16) * QDIM] = acc1[t][e] + bias;
        }
    }
}

// ---------------------------------------------------------------------------
// Launch
// ---------------------------------------------------------------------------
extern "C" void kernel_launch(void* const* d_in, const int* in_sizes, int n_in,
                              void* d_out, int out_size, void* d_ws, size_t ws_size,
                              hipStream_t stream) {
    const float* x   = (const float*)d_in[0];
    const float* ctx = (const float*)d_in[1];
    const float* Wq  = (const float*)d_in[2];
    const float* Wk  = (const float*)d_in[3];
    const float* Wv  = (const float*)d_in[4];
    const float* Wo  = (const float*)d_in[5];
    const float* bo  = (const float*)d_in[6];
    float* out = (float*)d_out;

    char* ws = (char*)d_ws;
    bf16* qb   = (bf16*)(ws + (size_t)0);
    bf16* kb   = (bf16*)(ws + ((size_t)4  << 20));
    bf16* vtb  = (bf16*)(ws + ((size_t)8  << 20));
    bf16* attO = (bf16*)(ws + ((size_t)12 << 20));
    bf16* WqT  = (bf16*)(ws + ((size_t)16 << 20));
    bf16* WkT  = (bf16*)(ws + ((size_t)17 << 20));
    bf16* WvT  = (bf16*)(ws + ((size_t)18 << 20));
    bf16* WoT  = (bf16*)(ws + ((size_t)19 << 20));

    const int wElems = QDIM * INNER;  // 524288
    // Wq/Wk/Wv: 1024 x 512 (lgC=9); Wo: 512 x 1024 (lgC=10)
    transpose_cvt<<<(wElems + 255) / 256, 256, 0, stream>>>(Wq, WqT, QDIM, 9);
    transpose_cvt<<<(wElems + 255) / 256, 256, 0, stream>>>(Wk, WkT, CDIM, 9);
    transpose_cvt<<<(wElems + 255) / 256, 256, 0, stream>>>(Wv, WvT, CDIM, 9);
    transpose_cvt<<<(wElems + 255) / 256, 256, 0, stream>>>(Wo, WoT, INNER, 10);

    // rows = B*N = 4096 -> 16 blocks of 256 rows; cols 512 -> 8 tiles of 64
    qkv_gemm<<<dim3(16, 8, 3), 256, 0, stream>>>(x, ctx, WqT, WkT, WvT, qb, kb, vtb);

    // 2048 queries / (8 waves * 16) = 16 blocks; heads 8; batch 2
    attn_kernel<<<dim3(16, HEADS, BATCH), 256, 0, stream>>>(qb, kb, vtb, attO);

    // rows 4096 -> 16 blocks; cols 1024 -> 16 tiles of 64
    out_gemm<<<dim3(16, 16, 1), 256, 0, stream>>>(attO, WoT, bo, out);
}
